// BiAttention_60730837565863
// MI455X (gfx1250) — matrix operand order repaired and checked by reading, not compiled, not run on hardware
//
#include <hip/hip_runtime.h>
#include <hip/hip_bf16.h>
#include <math.h>

#define B_ 8
#define L_ 2048
#define M_ 2048
#define D_ 256

typedef __attribute__((ext_vector_type(16))) __bf16 v16bf;
typedef __attribute__((ext_vector_type(8)))  float  v8f;

// reductions within a 16-lane half of a wave32 (xor offsets < 16 stay in the half)
__device__ __forceinline__ float half16_max(float v) {
#pragma unroll
    for (int off = 8; off > 0; off >>= 1) v = fmaxf(v, __shfl_xor(v, off, 32));
    return v;
}
__device__ __forceinline__ float half16_sum(float v) {
#pragma unroll
    for (int off = 8; off > 0; off >>= 1) v += __shfl_xor(v, off, 32);
    return v;
}

// ---------------- phase 0: input_dot[b,l], memory_dot[b,m] ----------------
__global__ __launch_bounds__(256)
void biatt_dots(const float* __restrict__ input, const float* __restrict__ memory,
                const float* __restrict__ w_input, const float* __restrict__ w_memory,
                float* __restrict__ idot, float* __restrict__ mdot) {
    const int gwave = (blockIdx.x * blockDim.x + threadIdx.x) >> 5;
    const int lane  = threadIdx.x & 31;
    const int nI = B_ * L_;
    const float* src; const float* w; float* dst; int row;
    if (gwave < nI) { row = gwave; src = input; w = w_input; dst = idot; }
    else {
        row = gwave - nI;
        if (row >= B_ * M_) return;
        src = memory; w = w_memory; dst = mdot;
    }
    const float* p = src + (size_t)row * D_;
    float s = 0.f;
#pragma unroll
    for (int j = 0; j < D_ / 32; ++j) s += p[lane + j * 32] * w[lane + j * 32];
#pragma unroll
    for (int off = 16; off > 0; off >>= 1) s += __shfl_xor(s, off, 32);
    if (lane == 0) dst[row] = s;
}

// ---------------- phase 1: fused flash cross-attention ----------------
// grid: B * (L/128) blocks, 256 threads (8 waves); each wave owns 16 L-rows.
__global__ __launch_bounds__(256)
void biatt_flash(const float* __restrict__ input, const float* __restrict__ memory,
                 const float* __restrict__ mask, const float* __restrict__ dot_scale,
                 const float* __restrict__ idot, const float* __restrict__ mdot,
                 float* __restrict__ out /*[B,L,4D], fills [D,2D)*/,
                 float* __restrict__ rmax /*[B,L]*/) {
    __shared__ __bf16 smem_mem [32][D_ + 8];   // staged memory tile, [m][d] (bf16)
    __shared__ __bf16 smem_memT[D_][32 + 2];   // transposed copy, [d][m] (bf16)
    __shared__ __bf16 smem_p[8][16][32 + 4];   // per-wave P transpose scratch

    const int wave = threadIdx.x >> 5;
    const int lane = threadIdx.x & 31;
    const int half = lane >> 4;       // which 16-lane half
    const int col  = lane & 15;

    const int b     = blockIdx.x / (L_ / 128);
    const int ltile = blockIdx.x % (L_ / 128);
    const int l0    = ltile * 128 + wave * 16;

    // A fragments: rows l0..l0+15 of (input * dot_scale), K = D in 8 chunks of 32.
    // A k order used here: elem e -> K = half*16 + e (contiguous, same as B)
    v16bf afrag[8];
    {
        const float* ip = input + ((size_t)(b * L_ + l0 + col)) * D_;
#pragma unroll
        for (int kk = 0; kk < 8; ++kk) {
            const int kb = kk * 32 + half * 16;
#pragma unroll
            for (int j = 0; j < 8; ++j) {
                afrag[kk][j]     = (__bf16)(ip[kb + j]      * dot_scale[kb + j]);
                afrag[kk][8 + j] = (__bf16)(ip[kb + 8 + j] * dot_scale[kb + 8 + j]);
            }
        }
    }

    // per-lane row bias (C layout: elem r -> row r + 8*half)
    float idotv[8];
#pragma unroll
    for (int r = 0; r < 8; ++r) idotv[r] = idot[b * L_ + l0 + r + 8 * half];

    float mrun[8], srun[8];
#pragma unroll
    for (int r = 0; r < 8; ++r) { mrun[r] = -3.0e38f; srun[r] = 0.0f; }
    v8f acc[16];
#pragma unroll
    for (int dt = 0; dt < 16; ++dt) acc[dt] = {};

    for (int m0 = 0; m0 < M_; m0 += 32) {
        // --- cooperative staging: 32 x 256 f32 -> bf16 in LDS, both layouts
        {
            const int mrow   = threadIdx.x >> 3;
            const int dstart = (threadIdx.x & 7) * 32;
            const float* mp  = memory + ((size_t)(b * M_ + m0 + mrow)) * D_ + dstart;
#pragma unroll
            for (int j = 0; j < 32; ++j) {
                const __bf16 v = (__bf16)mp[j];
                smem_mem [mrow][dstart + j] = v;
                smem_memT[dstart + j][mrow] = v;
            }
            if (m0 + 32 < M_) __builtin_prefetch(mp + 32 * D_, 0, 0);  // next tile
        }
        __syncthreads();

        // --- score GEMM: preload all B fragments, then one WMMA burst
        // B k order used here (same as A): elem e -> K = half*16 + e (within K-chunk), lane col = N
        v16bf bs0[8], bs1[8];
#pragma unroll
        for (int kk = 0; kk < 8; ++kk) {
            const __bf16* r0 = &smem_mem[col][kk * 32 + half * 16];
            const __bf16* r1 = &smem_mem[16 + col][kk * 32 + half * 16];
#pragma unroll
            for (int e = 0; e < 16; ++e) { bs0[kk][e] = r0[e]; bs1[kk][e] = r1[e]; }
        }
        v8f c0 = {}, c1 = {};
#pragma unroll
        for (int kk = 0; kk < 8; ++kk) {
            c0 = __builtin_amdgcn_wmma_f32_16x16x32_bf16(false, afrag[kk], false, bs0[kk], (short)0, c0, false, false);
            c1 = __builtin_amdgcn_wmma_f32_16x16x32_bf16(false, afrag[kk], false, bs1[kk], (short)0, c1, false, false);
        }

        const float md0 = mdot[b * M_ + m0 + col];
        const float md1 = mdot[b * M_ + m0 + 16 + col];
        const float mk0 = -1.0e30f * (1.0f - mask[b * M_ + m0 + col]);
        const float mk1 = -1.0e30f * (1.0f - mask[b * M_ + m0 + 16 + col]);

        // --- online softmax per row, stash P (bf16) for the PV GEMM
#pragma unroll
        for (int r = 0; r < 8; ++r) {
            float v0 = c0[r] + idotv[r] + md0 + mk0;
            float v1 = c1[r] + idotv[r] + md1 + mk1;
            float tmax  = half16_max(fmaxf(v0, v1));
            float mnew  = fmaxf(mrun[r], tmax);
            float alpha = __expf(mrun[r] - mnew);
            float p0 = __expf(v0 - mnew);
            float p1 = __expf(v1 - mnew);
            float rs = half16_sum(p0 + p1);
            srun[r] = srun[r] * alpha + rs;
            mrun[r] = mnew;
#pragma unroll
            for (int dt = 0; dt < 16; ++dt) acc[dt][r] *= alpha;
            smem_p[wave][r + 8 * half][col]      = (__bf16)p0;
            smem_p[wave][r + 8 * half][16 + col] = (__bf16)p1;
        }
        asm volatile("s_wait_dscnt 0" ::: "memory");  // wave-local LDS RAW fence

        // P as A fragment (16 rows x K=32 of m_local)
        v16bf pa;
        {
            const __bf16* pr = &smem_p[wave][col][half * 16];
#pragma unroll
            for (int j = 0; j < 8; ++j) { pa[j] = pr[j]; pa[8 + j] = pr[8 + j]; }
        }

        // --- PV GEMM: preload all 16 B fragments from the transposed tile
        // B: lane col = N = d-in-tile, elem e -> K = m_local = half*16 + e (contiguous!)
        v16bf bv[16];
#pragma unroll
        for (int dt = 0; dt < 16; ++dt) {
            const __bf16* tp = &smem_memT[dt * 16 + col][half * 16];
#pragma unroll
            for (int e = 0; e < 16; ++e) bv[dt][e] = tp[e];
        }
#pragma unroll
        for (int dt = 0; dt < 16; ++dt)
            acc[dt] = __builtin_amdgcn_wmma_f32_16x16x32_bf16(false, pa, false, bv[dt], (short)0, acc[dt], false, false);

        __syncthreads();
    }

    // --- finalize: output_one into out[..., D:2D), raw row-max to ws
#pragma unroll
    for (int r = 0; r < 8; ++r) {
        const int gl = l0 + r + 8 * half;
        const float inv = 1.0f / srun[r];
        float* orow = out + ((size_t)(b * L_ + gl)) * (4 * D_) + D_;
#pragma unroll
        for (int dt = 0; dt < 16; ++dt) orow[dt * 16 + col] = acc[dt][r] * inv;
        if (col == 0) rmax[b * L_ + gl] = mrun[r];
    }
}

// ---------------- phase 2: softmax over L of rowmax, output_two ----------------
__global__ __launch_bounds__(256)
void biatt_phase2(const float* __restrict__ input, const float* __restrict__ rmax,
                  float* __restrict__ out2 /*[B,D]*/) {
    const int b = blockIdx.x, tid = threadIdx.x;
    __shared__ float red[256];
    float lm = -3.4e38f;
    for (int l = tid; l < L_; l += 256) lm = fmaxf(lm, rmax[b * L_ + l]);
    red[tid] = lm; __syncthreads();
    for (int s = 128; s > 0; s >>= 1) { if (tid < s) red[tid] = fmaxf(red[tid], red[tid + s]); __syncthreads(); }
    const float gmax = red[0]; __syncthreads();
    float ls = 0.f;
    for (int l = tid; l < L_; l += 256) ls += __expf(rmax[b * L_ + l] - gmax);
    red[tid] = ls; __syncthreads();
    for (int s = 128; s > 0; s >>= 1) { if (tid < s) red[tid] += red[tid + s]; __syncthreads(); }
    const float gsum = red[0];
    // thread tid owns feature d = tid
    float a = 0.f;
    for (int l = 0; l < L_; ++l)
        a += __expf(rmax[b * L_ + l] - gmax) * input[((size_t)(b * L_ + l)) * D_ + tid];
    out2[b * D_ + tid] = a / gsum;
}

// ---------------- phase 3: final concat assembly ----------------
__global__ __launch_bounds__(256)
void biatt_phase3(const float* __restrict__ input, const float* __restrict__ out2,
                  float* __restrict__ out) {
    const int idx = blockIdx.x * 256 + threadIdx.x;   // over B*L*D
    const int d  = idx & (D_ - 1);
    const int bl = idx >> 8;          // idx / D
    const int b  = bl >> 11;          // bl / L
    const float inp = input[idx];
    float* orow = out + (size_t)bl * (4 * D_);
    const float o1 = orow[D_ + d];
    const float o2 = out2[b * D_ + d];
    orow[d]           = inp;
    orow[2 * D_ + d]  = inp * o1;
    orow[3 * D_ + d]  = o2 * o1;
}

extern "C" void kernel_launch(void* const* d_in, const int* in_sizes, int n_in,
                              void* d_out, int out_size, void* d_ws, size_t ws_size,
                              hipStream_t stream) {
    const float* input     = (const float*)d_in[0];
    const float* memory    = (const float*)d_in[1];
    const float* mask      = (const float*)d_in[2];
    const float* w_input   = (const float*)d_in[3];
    const float* w_memory  = (const float*)d_in[4];
    const float* dot_scale = (const float*)d_in[5];
    float* out = (float*)d_out;

    float* ws   = (float*)d_ws;
    float* idot = ws;                      // B*L
    float* mdot = idot + B_ * L_;          // B*M
    float* rmax = mdot + B_ * M_;          // B*L
    float* out2 = rmax + B_ * L_;          // B*D

    dim3 blk(256);
    const int nrows = B_ * L_ + B_ * M_;           // one wave per row
    biatt_dots<<<(nrows + 7) / 8, blk, 0, stream>>>(input, memory, w_input, w_memory, idot, mdot);
    biatt_flash<<<B_ * (L_ / 128), blk, 0, stream>>>(input, memory, mask, dot_scale,
                                                     idot, mdot, out, rmax);
    biatt_phase2<<<B_, blk, 0, stream>>>(input, rmax, out2);
    biatt_phase3<<<(B_ * L_ * D_) / 256, blk, 0, stream>>>(input, out2, out);
}
